// BiAttention_87376814670072
// MI455X (gfx1250) — matrix operand order repaired and checked by reading, not compiled, not run on hardware
//
#include <hip/hip_runtime.h>

// BiAttention for MI455X (gfx1250). B=16, L1=L2=1024, D=512, 4D=2048.
// All four GEMM stages on v_wmma_f32_16x16x32_bf16, 64x128 block tiles,
// 32x32 per wave. attn staging via TDM (tensor_load_to_lds) with double
// buffering + s_wait_tensorcnt when the builtin is available.

typedef __bf16 bf16_t;
typedef __attribute__((ext_vector_type(16))) __bf16 v16bf;
typedef __attribute__((ext_vector_type(8)))  __bf16 v8bf;
typedef __attribute__((ext_vector_type(8)))  float  v8f;

#define LDK 40            // LDS row stride: 32 K + 8 pad bf16 (80B rows)
#define NEGV (-1.0e12f)

#define WMMA_BF16(A, Bv, C) \
  __builtin_amdgcn_wmma_f32_16x16x32_bf16(false, (A), false, (Bv), (short)0, (C), false, false)

#if __has_builtin(__builtin_amdgcn_tensor_load_to_lds)
#define USE_TDM 1
#else
#define USE_TDM 0
#endif

// ---- WMMA fragment loaders (ISA 7.12.2 layouts) ------------------------------
// A (16x32 bf16): lane m=lane&15; kh=8*(lane>>4). elems 0..7 = K kh..kh+7,
// elems 8..15 = K kh+16..kh+23.
__device__ __forceinline__ v16bf load_a_frag(const bf16_t* base) {
  const int lane = threadIdx.x & 31;
  const bf16_t* p = base + (lane & 15) * LDK + ((lane >> 4) << 3);
  v8bf lo = *(const v8bf*)(p);
  v8bf hi = *(const v8bf*)(p + 16);
  return __builtin_shufflevector(lo, hi, 0,1,2,3,4,5,6,7,8,9,10,11,12,13,14,15);
}
// B (32x16 bf16) staged transposed in LDS as [n][k]: lane n=lane&15;
// kh=8*(lane>>4): elems 0..7 = K kh..kh+7, elems 8..15 = K kh+16..kh+23.
__device__ __forceinline__ v16bf load_b_frag(const bf16_t* base) {
  const int lane = threadIdx.x & 31;
  const bf16_t* p = base + (lane & 15) * LDK + ((lane >> 4) << 3);
  v8bf lo = *(const v8bf*)(p);
  v8bf hi = *(const v8bf*)(p + 16);
  return __builtin_shufflevector(lo, hi, 0,1,2,3,4,5,6,7,8,9,10,11,12,13,14,15);
}

// One 32x32-per-wave K-step: 2 A frags x 2 B frags -> 4 WMMA.
__device__ __forceinline__ void mm_step(const bf16_t (*As)[LDK], const bf16_t (*Bs)[LDK],
                                        int wr, int wc, v8f acc[2][2]) {
  v16bf a0 = load_a_frag(&As[wr * 32][0]);
  v16bf a1 = load_a_frag(&As[wr * 32 + 16][0]);
  v16bf b0 = load_b_frag(&Bs[wc * 32][0]);
  v16bf b1 = load_b_frag(&Bs[wc * 32 + 16][0]);
  acc[0][0] = WMMA_BF16(a0, b0, acc[0][0]);
  acc[0][1] = WMMA_BF16(a0, b1, acc[0][1]);
  acc[1][0] = WMMA_BF16(a1, b0, acc[1][0]);
  acc[1][1] = WMMA_BF16(a1, b1, acc[1][1]);
}

#if USE_TDM
typedef __attribute__((ext_vector_type(4))) unsigned int v4u;
typedef __attribute__((ext_vector_type(8))) int v8i;
typedef __attribute__((ext_vector_type(4))) int v4i;

// 2D TDM tile load: tile_dim0=32 bf16 (contiguous, 64B rows), tile_dim1=rows,
// tensor row stride 512 bf16. Hardware LDS padding: 4 DWORDs after every
// 16 DWORDs -> matches LDK=40 row stride exactly. (D# per ISA 8.3/8.4.)
__device__ __forceinline__ void tdm_load_tile(void* lds_ptr, const bf16_t* gptr,
                                              unsigned rows) {
  unsigned long long ga = (unsigned long long)(uintptr_t)gptr;
  v4u g0;
  g0.x = 1u;                                              // count=1, user mode
  g0.y = (unsigned)(uintptr_t)lds_ptr;                    // lds_addr
  g0.z = (unsigned)ga;                                    // global_addr[31:0]
  g0.w = (unsigned)((ga >> 32) & 0x01ffffffu) | (2u << 30); // addr[56:32]|type=2
  v8i g1;
  g1[0] = (int)((1u << 16) | (1u << 20) | (3u << 22) | (3u << 25));
          // data_size=2B | pad_enable | pad_interval=16DW | pad_amount=4DW
  g1[1] = (int)(512u << 16);             // tensor_dim0[15:0]=512 -> bits[63:48]
  g1[2] = 0;                             // tensor_dim0[31:16]=0, tensor_dim1[15:0]=0
  g1[3] = (int)(16u | (32u << 16));      // tensor_dim1[31:16]=16 (dim1=1M), tile_dim0=32
  g1[4] = (int)rows;                     // tile_dim1=rows, tile_dim2=0
  g1[5] = 512;                           // tensor_dim0_stride[31:0]
  g1[6] = 0;
  g1[7] = 0;
  v4i z4 = {0, 0, 0, 0};
#if __clang_major__ >= 23
  v8i z8 = {0, 0, 0, 0, 0, 0, 0, 0};
  __builtin_amdgcn_tensor_load_to_lds(g0, g1, z4, z4, z8, 0);  // 6-arg (clang-23+)
#else
  __builtin_amdgcn_tensor_load_to_lds(g0, g1, z4, z4, 0);      // 5-arg (ROCm 7.2)
#endif
}
#endif

// ---- Kernel 1: per-row prep: bf16 convert, optional v-scaled copy, bias dot --
__global__ __launch_bounds__(256)
void prep_rows_kernel(const float* __restrict__ h, const float* __restrict__ w,
                      const float* __restrict__ vvec, bf16_t* __restrict__ hb,
                      bf16_t* __restrict__ hvb, float* __restrict__ r) {
  const int row = blockIdx.x;            // b*1024 + l
  const int t = threadIdx.x;
  const float* src = h + ((size_t)row << 9);
  float part = 0.0f;
  for (int d = t; d < 512; d += 256) {
    float x = src[d];
    hb[((size_t)row << 9) + d] = (bf16_t)x;
    if (hvb) hvb[((size_t)row << 9) + d] = (bf16_t)(x * vvec[d]);
    part += x * w[d];
  }
  __shared__ float red[256];
  red[t] = part; __syncthreads();
  for (int s = 128; s > 0; s >>= 1) { if (t < s) red[t] += red[t + s]; __syncthreads(); }
  if (t == 0) r[row] = red[0];
}

// ---- Kernel 2: fp32 -> bf16 weight convert -----------------------------------
__global__ __launch_bounds__(256)
void convert_w_kernel(const float* __restrict__ src, bf16_t* __restrict__ dst, int n) {
  int i = (blockIdx.x * 256 + threadIdx.x) * 4;
  if (i < n) {
    #pragma unroll
    for (int e = 0; e < 4; ++e) dst[i + e] = (bf16_t)src[i + e];
  }
}

// ---- Kernel 3: attn logits: (h1*v) @ h2^T + r1 + r2 + bias, masked -----------
// Block tile 64(L1) x 128(L2); 8 waves of 32x32.
__global__ __launch_bounds__(256)
void attn_kernel(const bf16_t* __restrict__ h1v, const bf16_t* __restrict__ h2b,
                 const float* __restrict__ r1, const float* __restrict__ r2,
                 const float* __restrict__ bias,
                 const unsigned char* __restrict__ mask1,
                 const unsigned char* __restrict__ mask2,
                 float* __restrict__ attn) {
  __shared__ bf16_t As[2][64][LDK];
  __shared__ bf16_t Bs[2][128][LDK];
  const int b  = blockIdx.z;
  const int m0 = blockIdx.y * 64;       // L1 tile
  const int n0 = blockIdx.x * 128;      // L2 tile
  const int t = threadIdx.x;
  const int wave = t >> 5, lane = t & 31;
  const int wr = wave & 1;              // M sub-tile (32 rows)
  const int wc = wave >> 1;             // N sub-tile (32 cols)
  const bf16_t* aBase = h1v + ((size_t)((b << 10) + m0) << 9);
  const bf16_t* bBase = h2b + ((size_t)((b << 10) + n0) << 9);
  v8f acc[2][2] = {};
#if USE_TDM
  if (wave == 0) {
    tdm_load_tile(&As[0][0][0], aBase, 64);
    tdm_load_tile(&Bs[0][0][0], bBase, 128);
  }
  for (int i = 0; i < 16; ++i) {
    const int cur = i & 1;
    if (wave == 0) {
      if (i + 1 < 16) {
        tdm_load_tile(&As[cur ^ 1][0][0], aBase + (i + 1) * 32, 64);
        tdm_load_tile(&Bs[cur ^ 1][0][0], bBase + (i + 1) * 32, 128);
        __builtin_amdgcn_s_wait_tensorcnt(2);   // tile i complete, i+1 in flight
      } else {
        __builtin_amdgcn_s_wait_tensorcnt(0);
      }
    }
    __syncthreads();
    mm_step(As[cur], Bs[cur], wr, wc, acc);
    __syncthreads();
  }
#else
  const int ar = t >> 2, ak = (t & 3) * 8;   // A staging: 64 rows x 32 k
  const int br = t >> 1, bk2 = (t & 1) * 16; // B staging: 128 rows x 32 k
  for (int k0 = 0; k0 < 512; k0 += 32) {
    *(v8bf*)&As[0][ar][ak] = *(const v8bf*)(aBase + ((size_t)ar << 9) + k0 + ak);
    *(v8bf*)&Bs[0][br][bk2] = *(const v8bf*)(bBase + ((size_t)br << 9) + k0 + bk2);
    *(v8bf*)&Bs[0][br][bk2 + 8] = *(const v8bf*)(bBase + ((size_t)br << 9) + k0 + bk2 + 8);
    if (k0 + 32 < 512) {
      __builtin_prefetch(aBase + ((size_t)ar << 9) + k0 + 32 + ak, 0, 0);
      __builtin_prefetch(bBase + ((size_t)br << 9) + k0 + 32 + bk2, 0, 0);
    }
    __syncthreads();
    mm_step(As[0], Bs[0], wr, wc, acc);
    __syncthreads();
  }
#endif
  const float sb = bias[0];
  const int nlane = lane & 15, mh = (lane >> 4) * 8;
  #pragma unroll
  for (int jm = 0; jm < 2; ++jm) {
    #pragma unroll
    for (int jn = 0; jn < 2; ++jn) {
      const int nn = n0 + wc * 32 + jn * 16 + nlane;
      const float cb = r2[(b << 10) + nn] + sb;
      const bool mN = mask2[(b << 10) + nn] != 0;
      #pragma unroll
      for (int i = 0; i < 8; ++i) {
        const int mm = m0 + wr * 32 + jm * 16 + mh + i;
        float x = acc[jm][jn][i] + r1[(b << 10) + mm] + cb;
        if (mN || mask1[(b << 10) + mm]) x = NEGV;
        attn[(((size_t)((b << 10) + mm)) << 10) + nn] = x;
      }
    }
  }
}

// ---- Kernel 4: row softmax stats (over axis 2) -------------------------------
__global__ __launch_bounds__(256)
void row_stats_kernel(const float* __restrict__ attn, float* __restrict__ rmax,
                      float* __restrict__ rsum) {
  const size_t row = blockIdx.x;
  const float* p = attn + (row << 10);
  const int t = threadIdx.x;
  __shared__ float red[256];
  float mx = -3.4e38f;
  for (int j = t; j < 1024; j += 256) mx = fmaxf(mx, p[j]);
  red[t] = mx; __syncthreads();
  for (int s = 128; s > 0; s >>= 1) { if (t < s) red[t] = fmaxf(red[t], red[t + s]); __syncthreads(); }
  mx = red[0]; __syncthreads();
  float sm = 0.0f;
  for (int j = t; j < 1024; j += 256) sm += __expf(p[j] - mx);
  red[t] = sm; __syncthreads();
  for (int s = 128; s > 0; s >>= 1) { if (t < s) red[t] += red[t + s]; __syncthreads(); }
  if (t == 0) { rmax[row] = mx; rsum[row] = red[0]; }
}

// ---- Kernel 5: col softmax stats (over axis 1) -------------------------------
__global__ __launch_bounds__(256)
void col_stats_kernel(const float* __restrict__ attn, float* __restrict__ cmax,
                      float* __restrict__ csum) {
  const int b = blockIdx.x >> 2;
  const int m = ((blockIdx.x & 3) << 8) + threadIdx.x;
  const float* p = attn + ((size_t)b << 20) + m;
  float mx = -3.4e38f;
  for (int l = 0; l < 1024; ++l) mx = fmaxf(mx, p[(size_t)l << 10]);
  float sm = 0.0f;
  for (int l = 0; l < 1024; ++l) sm += __expf(p[(size_t)l << 10] - mx);
  cmax[(b << 10) + m] = mx;
  csum[(b << 10) + m] = sm;
}

// ---- Kernel 6: pooled summary: pool[b,d] = sum_l softmax(stat[b,:])[l]*h[b,l,d]
__global__ __launch_bounds__(256)
void pool_kernel(const float* __restrict__ stat, const float* __restrict__ h,
                 float* __restrict__ pool) {
  const int b = blockIdx.x;
  const int t = threadIdx.x;
  const float* s = stat + (b << 10);
  __shared__ float p[1024];
  __shared__ float red[256];
  float mx = -3.4e38f;
  for (int l = t; l < 1024; l += 256) mx = fmaxf(mx, s[l]);
  red[t] = mx; __syncthreads();
  for (int sh = 128; sh > 0; sh >>= 1) { if (t < sh) red[t] = fmaxf(red[t], red[t + sh]); __syncthreads(); }
  mx = red[0]; __syncthreads();
  float sm = 0.0f;
  for (int l = t; l < 1024; l += 256) { float e = __expf(s[l] - mx); p[l] = e; sm += e; }
  red[t] = sm; __syncthreads();
  for (int sh = 128; sh > 0; sh >>= 1) { if (t < sh) red[t] += red[t + sh]; __syncthreads(); }
  const float inv = 1.0f / red[0];
  __syncthreads();
  for (int d = t; d < 512; d += 256) {
    const float* hp = h + ((size_t)b << 19) + d;
    float acc = 0.0f;
    for (int l = 0; l < 1024; ++l) acc += p[l] * hp[(size_t)l << 9];
    pool[(b << 9) + d] = acc * inv;
  }
}

// ---- Kernel 7: A21[b,l,d] = softmax_axis2(attn) @ h2 -------------------------
// Block tile 64(l) x 128(d), K = L2 = 1024. Softmax fused into A staging.
__global__ __launch_bounds__(256)
void a21_kernel(const float* __restrict__ attn, const float* __restrict__ rmax,
                const float* __restrict__ rsum, const bf16_t* __restrict__ hb,
                float* __restrict__ out) {
  __shared__ bf16_t As[64][LDK];
  __shared__ bf16_t Bs[128][LDK];
  const int b  = blockIdx.z;
  const int r0 = blockIdx.y * 64;       // l tile
  const int n0 = blockIdx.x * 128;      // d tile
  const int t = threadIdx.x;
  const int wave = t >> 5, lane = t & 31;
  const int wr = wave & 1, wc = wave >> 1;
  const int ar = t >> 2, ak = (t & 3) * 8;   // A: fixed row, 8 k's
  const int bk = t >> 3, bn = (t & 7) * 8;   // B: fixed k, 8 n's (x2 halves)
  const float rm = rmax[(b << 10) + r0 + ar];
  const float ri = 1.0f / rsum[(b << 10) + r0 + ar];
  const float* ap = attn + (((size_t)((b << 10) + r0 + ar)) << 10) + ak;
  const bf16_t* bp = hb + (((size_t)((b << 10) + bk)) << 9) + n0 + bn;
  v8f acc[2][2] = {};
  for (int k0 = 0; k0 < 1024; k0 += 32) {
    v8bf av;
    #pragma unroll
    for (int e = 0; e < 8; ++e) av[e] = (bf16_t)(__expf(ap[k0 + e] - rm) * ri);
    *(v8bf*)&As[ar][ak] = av;
    v8bf bv0 = *(const v8bf*)(bp + ((size_t)k0 << 9));
    v8bf bv1 = *(const v8bf*)(bp + ((size_t)k0 << 9) + 64);
    #pragma unroll
    for (int e = 0; e < 8; ++e) Bs[bn + e][bk] = bv0[e];
    #pragma unroll
    for (int e = 0; e < 8; ++e) Bs[bn + 64 + e][bk] = bv1[e];
    __syncthreads();
    mm_step(As, Bs, wr, wc, acc);
    __syncthreads();
  }
  const int nlane = lane & 15, mh = (lane >> 4) * 8;
  #pragma unroll
  for (int jm = 0; jm < 2; ++jm)
    #pragma unroll
    for (int jn = 0; jn < 2; ++jn) {
      const int nn = n0 + wc * 32 + jn * 16 + nlane;
      #pragma unroll
      for (int i = 0; i < 8; ++i)
        out[(((size_t)((b << 10) + r0 + wr * 32 + jm * 16 + mh + i)) << 9) + nn] =
            acc[jm][jn][i];
    }
}

// ---- Kernel 8: A12[b,m,d] = softmax_axis1(attn)^T @ h1 -----------------------
__global__ __launch_bounds__(256)
void a12_kernel(const float* __restrict__ attn, const float* __restrict__ cmax,
                const float* __restrict__ csum, const bf16_t* __restrict__ hb,
                float* __restrict__ out) {
  __shared__ bf16_t As[64][LDK];
  __shared__ bf16_t Bs[128][LDK];
  const int b  = blockIdx.z;
  const int r0 = blockIdx.y * 64;       // m tile
  const int n0 = blockIdx.x * 128;      // d tile
  const int t = threadIdx.x;
  const int wave = t >> 5, lane = t & 31;
  const int wr = wave & 1, wc = wave >> 1;
  const int ar = t >> 2, ak = (t & 3) * 8;
  const int bk = t >> 3, bn = (t & 7) * 8;
  const float cm = cmax[(b << 10) + r0 + ar];
  const float ci = 1.0f / csum[(b << 10) + r0 + ar];
  const float* ap = attn + ((size_t)b << 20) + r0 + ar;        // + l*1024
  const bf16_t* bp = hb + (((size_t)((b << 10) + bk)) << 9) + n0 + bn;
  v8f acc[2][2] = {};
  for (int k0 = 0; k0 < 1024; k0 += 32) {
    v8bf av;
    #pragma unroll
    for (int e = 0; e < 8; ++e)
      av[e] = (bf16_t)(__expf(ap[((size_t)(k0 + ak + e)) << 10] - cm) * ci);
    *(v8bf*)&As[ar][ak] = av;
    v8bf bv0 = *(const v8bf*)(bp + ((size_t)k0 << 9));
    v8bf bv1 = *(const v8bf*)(bp + ((size_t)k0 << 9) + 64);
    #pragma unroll
    for (int e = 0; e < 8; ++e) Bs[bn + e][bk] = bv0[e];
    #pragma unroll
    for (int e = 0; e < 8; ++e) Bs[bn + 64 + e][bk] = bv1[e];
    __syncthreads();
    mm_step(As, Bs, wr, wc, acc);
    __syncthreads();
  }
  const int nlane = lane & 15, mh = (lane >> 4) * 8;
  #pragma unroll
  for (int jm = 0; jm < 2; ++jm)
    #pragma unroll
    for (int jn = 0; jn < 2; ++jn) {
      const int nn = n0 + wc * 32 + jn * 16 + nlane;
      #pragma unroll
      for (int i = 0; i < 8; ++i)
        out[(((size_t)((b << 10) + r0 + wr * 32 + jm * 16 + mh + i)) << 9) + nn] =
            acc[jm][jn][i];
    }
}

// ---- Kernel 9: merged = relu(cat(h, att, h*att, h*pool) @ W + bias) ----------
// K = 2048, concat materialized on the fly in A staging.
__global__ __launch_bounds__(256)
void merged_kernel(const float* __restrict__ h, const float* __restrict__ att,
                   const float* __restrict__ pool, const bf16_t* __restrict__ Wb,
                   const float* __restrict__ bias, float* __restrict__ out) {
  __shared__ bf16_t As[64][LDK];
  __shared__ bf16_t Bs[128][LDK];
  const int b  = blockIdx.z;
  const int r0 = blockIdx.y * 64;       // row tile (l or m)
  const int n0 = blockIdx.x * 128;      // out-feature tile
  const int t = threadIdx.x;
  const int wave = t >> 5, lane = t & 31;
  const int wr = wave & 1, wc = wave >> 1;
  const int ar = t >> 2, ak = (t & 3) * 8;
  const int bk = t >> 3, bn = (t & 7) * 8;
  const size_t rowbase = ((size_t)((b << 10) + r0 + ar)) << 9;
  v8f acc[2][2] = {};
  for (int k0 = 0; k0 < 2048; k0 += 32) {
    const int kk = k0 + ak;
    const int g = kk >> 9, j = kk & 511;
    const float* hp = h + rowbase + j;
    const float* tp = att + rowbase + j;
    v8bf av;
    if (g == 0) {
      #pragma unroll
      for (int e = 0; e < 8; ++e) av[e] = (bf16_t)hp[e];
    } else if (g == 1) {
      #pragma unroll
      for (int e = 0; e < 8; ++e) av[e] = (bf16_t)tp[e];
    } else if (g == 2) {
      #pragma unroll
      for (int e = 0; e < 8; ++e) av[e] = (bf16_t)(hp[e] * tp[e]);
    } else {
      const float* pp = pool + (b << 9) + j;
      #pragma unroll
      for (int e = 0; e < 8; ++e) av[e] = (bf16_t)(hp[e] * pp[e]);
    }
    *(v8bf*)&As[ar][ak] = av;
    v8bf wv0 = *(const v8bf*)(Wb + (((size_t)(k0 + bk)) << 9) + n0 + bn);
    v8bf wv1 = *(const v8bf*)(Wb + (((size_t)(k0 + bk)) << 9) + n0 + bn + 64);
    #pragma unroll
    for (int e = 0; e < 8; ++e) Bs[bn + e][bk] = wv0[e];
    #pragma unroll
    for (int e = 0; e < 8; ++e) Bs[bn + 64 + e][bk] = wv1[e];
    __syncthreads();
    mm_step(As, Bs, wr, wc, acc);
    __syncthreads();
  }
  const int nlane = lane & 15, mh = (lane >> 4) * 8;
  #pragma unroll
  for (int jm = 0; jm < 2; ++jm)
    #pragma unroll
    for (int jn = 0; jn < 2; ++jn) {
      const int nn = n0 + wc * 32 + jn * 16 + nlane;
      const float bb = bias[nn];
      #pragma unroll
      for (int i = 0; i < 8; ++i) {
        float x = acc[jm][jn][i] + bb;
        x = x > 0.0f ? x : 0.0f;
        out[(((size_t)((b << 10) + r0 + wr * 32 + jm * 16 + mh + i)) << 9) + nn] = x;
      }
    }
}

// ---- Host-side orchestration -------------------------------------------------
extern "C" void kernel_launch(void* const* d_in, const int* in_sizes, int n_in,
                              void* d_out, int out_size, void* d_ws, size_t ws_size,
                              hipStream_t stream) {
  const float* h1 = (const float*)d_in[0];
  const unsigned char* mask1 = (const unsigned char*)d_in[1];
  const float* h2 = (const float*)d_in[2];
  const unsigned char* mask2 = (const unsigned char*)d_in[3];
  const float* w1 = (const float*)d_in[4];
  const float* w2 = (const float*)d_in[5];
  const float* vv = (const float*)d_in[6];
  const float* bias = (const float*)d_in[7];
  const float* W1 = (const float*)d_in[8];
  const float* b1 = (const float*)d_in[9];
  const float* W2 = (const float*)d_in[10];
  const float* b2 = (const float*)d_in[11];
  float* out1 = (float*)d_out;
  float* out2 = out1 + (size_t)16 * 1024 * 512;

  char* ws = (char*)d_ws;
  auto carve = [&](size_t bytes) -> char* {
    char* p = ws;
    ws += (bytes + 255) & ~(size_t)255;
    return p;
  };
  const size_t NR = (size_t)16 * 1024;
  bf16_t* h1b  = (bf16_t*)carve(NR * 512 * 2);
  bf16_t* h2b  = (bf16_t*)carve(NR * 512 * 2);
  bf16_t* h1vb = (bf16_t*)carve(NR * 512 * 2);
  bf16_t* W1b  = (bf16_t*)carve((size_t)2048 * 512 * 2);
  bf16_t* W2b  = (bf16_t*)carve((size_t)2048 * 512 * 2);
  float* r1   = (float*)carve(NR * 4);
  float* r2   = (float*)carve(NR * 4);
  float* rmax = (float*)carve(NR * 4);
  float* rsum = (float*)carve(NR * 4);
  float* cmax = (float*)carve(NR * 4);
  float* csum = (float*)carve(NR * 4);
  float* pool1 = (float*)carve((size_t)16 * 512 * 4);
  float* pool2 = (float*)carve((size_t)16 * 512 * 4);
  float* A21  = (float*)carve(NR * 512 * 4);
  float* A12  = (float*)carve(NR * 512 * 4);
  float* attn = (float*)carve((size_t)16 * 1024 * 1024 * 4);

  // 1) prep: bf16 conversions + row-bias dot products
  prep_rows_kernel<<<16384, 256, 0, stream>>>(h1, w1, vv, h1b, h1vb, r1);
  prep_rows_kernel<<<16384, 256, 0, stream>>>(h2, w2, nullptr, h2b, nullptr, r2);
  convert_w_kernel<<<(2048 * 512 / 4 + 255) / 256, 256, 0, stream>>>(W1, W1b, 2048 * 512);
  convert_w_kernel<<<(2048 * 512 / 4 + 255) / 256, 256, 0, stream>>>(W2, W2b, 2048 * 512);

  // 2) attention logits (WMMA; TDM-staged when available)
  attn_kernel<<<dim3(8, 16, 16), 256, 0, stream>>>(h1vb, h2b, r1, r2, bias, mask1, mask2, attn);

  // 3) softmax stats + pooled summaries
  row_stats_kernel<<<16384, 256, 0, stream>>>(attn, rmax, rsum);
  col_stats_kernel<<<64, 256, 0, stream>>>(attn, cmax, csum);
  pool_kernel<<<16, 256, 0, stream>>>(rmax, h1, pool1);
  pool_kernel<<<16, 256, 0, stream>>>(cmax, h2, pool2);

  // 4) attention-weighted sums (WMMA, softmax fused into A staging)
  a21_kernel<<<dim3(4, 16, 16), 256, 0, stream>>>(attn, rmax, rsum, h2b, A21);
  a12_kernel<<<dim3(4, 16, 16), 256, 0, stream>>>(attn, cmax, csum, h1b, A12);

  // 5) merged projections (WMMA, concat + bias + relu fused)
  merged_kernel<<<dim3(4, 16, 16), 256, 0, stream>>>(h1, A21, pool1, W1b, b1, out1);
  merged_kernel<<<dim3(4, 16, 16), 256, 0, stream>>>(h2, A12, pool2, W2b, b2, out2);
}
